// RBF_58385785421876
// MI455X (gfx1250) — matrix operand order repaired and checked by reading, not compiled, not run on hardware
//
#include <hip/hip_runtime.h>
#include <hip/hip_bf16.h>

// ---- types ----
typedef __bf16 bf16;
typedef __attribute__((ext_vector_type(4)))  __bf16 v4bf;
typedef __attribute__((ext_vector_type(8)))  __bf16 v8bf;
typedef __attribute__((ext_vector_type(16))) __bf16 v16bf;
typedef __attribute__((ext_vector_type(8)))  float  v8f;

// problem sizes
#define BB 4096
#define II 1024
#define CC 2048
#define OO 1024

// ---------------------------------------------------------------------------
// CDNA5 async global->LDS copy (16B per lane), tracked with ASYNCcnt.
// Generic pointers to LDS carry the LDS byte offset in their low 32 bits.
// ---------------------------------------------------------------------------
__device__ __forceinline__ void async_copy_b128(const void* g, void* l) {
  unsigned int loff = (unsigned int)(unsigned long long)l;
  asm volatile("global_load_async_to_lds_b128 %0, %1, off"
               :
               : "v"(loff), "v"(g)
               : "memory");
}
__device__ __forceinline__ void wait_async_8() {
  asm volatile("s_wait_asynccnt 0x8" ::: "memory");
}
__device__ __forceinline__ void wait_async_0() {
  asm volatile("s_wait_asynccnt 0x0" ::: "memory");
}

// ---------------------------------------------------------------------------
// 1) BatchNorm statistics: per-column mean/var over 4096 rows -> scale/shift
// ---------------------------------------------------------------------------
__global__ __launch_bounds__(256) void bn_stats_kernel(
    const float* __restrict__ x, const float* __restrict__ gamma,
    const float* __restrict__ beta, float* __restrict__ scale,
    float* __restrict__ shift) {
  int col = blockIdx.x * 256 + threadIdx.x;  // I = 1024, 4 blocks
  float s = 0.f, s2 = 0.f;
  for (int row = 0; row < BB; ++row) {
    float v = x[(size_t)row * II + col];
    s += v;
    s2 += v * v;
  }
  float mean = s * (1.0f / BB);
  float var = s2 * (1.0f / BB) - mean * mean;
  float sc = gamma[col] * rsqrtf(var + 1e-5f);
  scale[col] = sc;
  shift[col] = beta[col] - mean * sc;
}

// ---------------------------------------------------------------------------
// 2) Normalize rows -> bf16 A matrix + per-row sum of squares
// ---------------------------------------------------------------------------
__global__ __launch_bounds__(256) void normalize_kernel(
    const float* __restrict__ x, const float* __restrict__ scale,
    const float* __restrict__ shift, bf16* __restrict__ xn,
    float* __restrict__ x_sq) {
  __shared__ float red[256];
  int m = blockIdx.x;          // one row per block
  int t = threadIdx.x;         // 256 threads * 4 elems = 1024
  const float4 xv = *(const float4*)(x + (size_t)m * II + t * 4);
  const float4 sc = *(const float4*)(scale + t * 4);
  const float4 sh = *(const float4*)(shift + t * 4);
  float a0 = xv.x * sc.x + sh.x;
  float a1 = xv.y * sc.y + sh.y;
  float a2 = xv.z * sc.z + sh.z;
  float a3 = xv.w * sc.w + sh.w;
  v4bf o;
  o[0] = (bf16)a0; o[1] = (bf16)a1; o[2] = (bf16)a2; o[3] = (bf16)a3;
  *(v4bf*)(xn + (size_t)m * II + t * 4) = o;
  red[t] = a0 * a0 + a1 * a1 + a2 * a2 + a3 * a3;
  __syncthreads();
  for (int off = 128; off > 0; off >>= 1) {
    if (t < off) red[t] += red[t + off];
    __syncthreads();
  }
  if (t == 0) x_sq[m] = red[0];
}

// ---------------------------------------------------------------------------
// 3a) per-center squared norm (column reduction over centers (I x C))
// ---------------------------------------------------------------------------
__global__ __launch_bounds__(256) void csq_kernel(
    const float* __restrict__ centers, float* __restrict__ c_sq) {
  int c = blockIdx.x * 256 + threadIdx.x;  // C = 2048, 8 blocks
  float s = 0.f;
  for (int i = 0; i < II; ++i) {
    float v = centers[(size_t)i * CC + c];
    s += v * v;
  }
  c_sq[c] = s;
}

// ---------------------------------------------------------------------------
// 3b) centers (I x C) -> centersT bf16 (C x I): N-major, K-contiguous for B op
// ---------------------------------------------------------------------------
__global__ __launch_bounds__(256) void transpose_centers_kernel(
    const float* __restrict__ centers, bf16* __restrict__ ctT) {
  __shared__ float tile[32][33];
  int c0 = blockIdx.x * 32;  // over C
  int i0 = blockIdx.y * 32;  // over I
  int tc = threadIdx.x & 31;
  int tr = threadIdx.x >> 5;  // 0..7
#pragma unroll
  for (int rr = 0; rr < 4; ++rr) {
    int i = i0 + tr + rr * 8;
    tile[tr + rr * 8][tc] = centers[(size_t)i * CC + c0 + tc];
  }
  __syncthreads();
#pragma unroll
  for (int rr = 0; rr < 4; ++rr) {
    int c = c0 + tr + rr * 8;
    ctT[(size_t)c * II + i0 + tc] = (bf16)tile[tc][tr + rr * 8];
  }
}

// ---------------------------------------------------------------------------
// 4) cast W (O x C, row-major == N-major K-contiguous) to bf16
// ---------------------------------------------------------------------------
__global__ __launch_bounds__(256) void castW_kernel(
    const float* __restrict__ W, bf16* __restrict__ Wb) {
  size_t idx = ((size_t)blockIdx.x * 256 + threadIdx.x) * 4;
  float4 v = *(const float4*)(W + idx);
  v4bf o;
  o[0] = (bf16)v.x; o[1] = (bf16)v.y; o[2] = (bf16)v.z; o[3] = (bf16)v.w;
  *(v4bf*)(Wb + idx) = o;
}

// ---------------------------------------------------------------------------
// WMMA fragment helpers (ISA 7.12.2 layouts, wave32). LDS tiles are
// 128 rows x 64 K bf16, row stride 64 elements (128 B).
//   A 16x32 bf16: lane<16 -> row=lane, K {0..7,16..23}; lane>=16 -> K {8..15,24..31}
//   B 32x16 bf16: lane<16 -> col=lane, K {0..7,16..23}; lane>=16 -> K {8..15,24..31}
// ---------------------------------------------------------------------------
__device__ __forceinline__ v16bf load_afrag(const bf16* As, int row_base,
                                            int ksub, int lane) {
  int r = lane & 15, h = lane >> 4;
  const bf16* p = As + (row_base + r) * 64 + ksub * 32 + h * 8;
  v8bf lo = *(const v8bf*)p;
  v8bf hi = *(const v8bf*)(p + 16);
  return __builtin_shufflevector(lo, hi, 0, 1, 2, 3, 4, 5, 6, 7, 8, 9, 10, 11,
                                 12, 13, 14, 15);
}

__device__ __forceinline__ v16bf load_bfrag(const bf16* Bs, int col_base,
                                            int ksub, int lane) {
  int r = lane & 15, h = lane >> 4;
  const bf16* p = Bs + (col_base + r) * 64 + ksub * 32 + h * 8;
  v8bf lo = *(const v8bf*)p;
  v8bf hi = *(const v8bf*)(p + 16);
  return __builtin_shufflevector(lo, hi, 0, 1, 2, 3, 4, 5, 6, 7, 8, 9, 10, 11,
                                 12, 13, 14, 15);
}

// stage one 128x64 bf16 tile pair (A and B) into an LDS buffer via async
// copies: 8 async instructions per thread (4 for A, 4 for B).
__device__ __forceinline__ void stage_tiles(const bf16* __restrict__ A,
                                            const bf16* __restrict__ Bm,
                                            int lda, int ldb, int m0, int n0,
                                            int k0, bf16* As, bf16* Bs,
                                            int tid) {
#pragma unroll
  for (int u = 0; u < 4; ++u) {
    int id = tid + u * 256;        // 1024 chunks of 16B per matrix
    int row = id >> 3, oc = id & 7;
    async_copy_b128(A + (size_t)(m0 + row) * lda + k0 + oc * 8,
                    As + row * 64 + oc * 8);
    async_copy_b128(Bm + (size_t)(n0 + row) * ldb + k0 + oc * 8,
                    Bs + row * 64 + oc * 8);
  }
}

// one 64-deep K step: 24 ds_load_b128 fragment loads + 16 WMMA
__device__ __forceinline__ void compute_tile(const bf16* As, const bf16* Bs,
                                             v8f (&acc)[2][4], int wm, int wn,
                                             int lane) {
#pragma unroll
  for (int ks = 0; ks < 2; ++ks) {
    v16bf af[2], bfr[4];
#pragma unroll
    for (int f = 0; f < 2; ++f)
      af[f] = load_afrag(As, wm * 32 + f * 16, ks, lane);
#pragma unroll
    for (int g = 0; g < 4; ++g)
      bfr[g] = load_bfrag(Bs, wn * 64 + g * 16, ks, lane);
#pragma unroll
    for (int f = 0; f < 2; ++f)
#pragma unroll
      for (int g = 0; g < 4; ++g)
        acc[f][g] = __builtin_amdgcn_wmma_f32_16x16x32_bf16(
            false, af[f], false, bfr[g], (short)0, acc[f][g], false, false);
  }
}

// ---------------------------------------------------------------------------
// software-pipelined GEMM mainloop: block tile 128x128, 8 waves (wave32) 4x2,
// wave tile 32x64, K-step 64, explicit ping-pong LDS buffers, async staging.
// nt (=K/64) must be even; tiles 0..nt-1, even->buf0, odd->buf1.
// ---------------------------------------------------------------------------
#define GEMM_MAINLOOP(Aptr, Bptr, LD, KTOT)                                    \
  stage_tiles(Aptr, Bptr, LD, LD, m0, n0, 0, As0, Bs0, tid);                   \
  {                                                                            \
    const int nt = (KTOT) / 64;                                                \
    _Pragma("unroll 1") for (int t = 0; t + 2 < nt; t += 2) {                  \
      stage_tiles(Aptr, Bptr, LD, LD, m0, n0, (t + 1) * 64, As1, Bs1, tid);    \
      wait_async_8();                                                          \
      __syncthreads();                                                         \
      compute_tile(As0, Bs0, acc, wm, wn, lane);                               \
      __syncthreads();                                                         \
      stage_tiles(Aptr, Bptr, LD, LD, m0, n0, (t + 2) * 64, As0, Bs0, tid);    \
      wait_async_8();                                                          \
      __syncthreads();                                                         \
      compute_tile(As1, Bs1, acc, wm, wn, lane);                               \
      __syncthreads();                                                         \
    }                                                                          \
    stage_tiles(Aptr, Bptr, LD, LD, m0, n0, (nt - 1) * 64, As1, Bs1, tid);     \
    wait_async_8();                                                            \
    __syncthreads();                                                           \
    compute_tile(As0, Bs0, acc, wm, wn, lane);                                 \
    __syncthreads();                                                           \
    wait_async_0();                                                            \
    __syncthreads();                                                           \
    compute_tile(As1, Bs1, acc, wm, wn, lane);                                 \
  }

// ---------------------------------------------------------------------------
// 5) GEMM1 + distance + phi epilogue:
//    acc = xn . centersT ; dist = x2[m]+c2[n]-2 acc ; phi = exp(-dist^2) (bf16)
//    M=4096, N=2048, K=1024.
// ---------------------------------------------------------------------------
__global__ __launch_bounds__(256) void gemm_phi_kernel(
    const bf16* __restrict__ A, const bf16* __restrict__ Bm,
    const float* __restrict__ x_sq, const float* __restrict__ c_sq,
    bf16* __restrict__ phi) {
  __shared__ __align__(16) bf16 As[2][128 * 64];
  __shared__ __align__(16) bf16 Bs[2][128 * 64];
  bf16* As0 = As[0];
  bf16* As1 = As[1];
  bf16* Bs0 = Bs[0];
  bf16* Bs1 = Bs[1];
  const int tid = threadIdx.x;
  const int lane = tid & 31;
  const int wave = tid >> 5;
  const int wm = wave >> 1;  // 0..3
  const int wn = wave & 1;   // 0..1
  const int m0 = blockIdx.y * 128;
  const int n0 = blockIdx.x * 128;

  v8f acc[2][4] = {};

  GEMM_MAINLOOP(A, Bm, II, II)

  // epilogue
  const int r0 = lane & 15, h = lane >> 4;
#pragma unroll
  for (int f = 0; f < 2; ++f) {
    int mb = m0 + wm * 32 + f * 16 + h * 8;
#pragma unroll
    for (int g = 0; g < 4; ++g) {
      int n = n0 + wn * 64 + g * 16 + r0;
      float cs = c_sq[n];
#pragma unroll
      for (int rr = 0; rr < 8; ++rr) {
        int m = mb + rr;
        float d = x_sq[m] + cs - 2.0f * acc[f][g][rr];
        float p = __expf(-d * d);
        phi[(size_t)m * CC + n] = (bf16)p;
      }
    }
  }
}

// ---------------------------------------------------------------------------
// 6) GEMM2 + bias: out = phi . W^T + b ; M=4096, N=1024, K=2048
// ---------------------------------------------------------------------------
__global__ __launch_bounds__(256) void gemm_out_kernel(
    const bf16* __restrict__ A, const bf16* __restrict__ Bm,
    const float* __restrict__ bias, float* __restrict__ out) {
  __shared__ __align__(16) bf16 As[2][128 * 64];
  __shared__ __align__(16) bf16 Bs[2][128 * 64];
  bf16* As0 = As[0];
  bf16* As1 = As[1];
  bf16* Bs0 = Bs[0];
  bf16* Bs1 = Bs[1];
  const int tid = threadIdx.x;
  const int lane = tid & 31;
  const int wave = tid >> 5;
  const int wm = wave >> 1;
  const int wn = wave & 1;
  const int m0 = blockIdx.y * 128;
  const int n0 = blockIdx.x * 128;

  v8f acc[2][4] = {};

  GEMM_MAINLOOP(A, Bm, CC, CC)

  const int r0 = lane & 15, h = lane >> 4;
#pragma unroll
  for (int f = 0; f < 2; ++f) {
    int mb = m0 + wm * 32 + f * 16 + h * 8;
#pragma unroll
    for (int g = 0; g < 4; ++g) {
      int o = n0 + wn * 64 + g * 16 + r0;
      float bv = bias[o];
#pragma unroll
      for (int rr = 0; rr < 8; ++rr) {
        int m = mb + rr;
        out[(size_t)m * OO + o] = acc[f][g][rr] + bv;
      }
    }
  }
}

// ---------------------------------------------------------------------------
// launch: ws layout (bytes):
//   0       scale   f32[1024]
//   4K      shift   f32[1024]
//   8K      x_sq    f32[4096]
//   24K     c_sq    f32[2048]
//   64K     xn      bf16[4096*1024]   (8 MB)
//   64K+8M  ctT     bf16[2048*1024]   (4 MB)
//   64K+12M Wb      bf16[1024*2048]   (4 MB)
//   64K+16M phi     bf16[4096*2048]   (16 MB)
// total ~32.06 MB
// ---------------------------------------------------------------------------
extern "C" void kernel_launch(void* const* d_in, const int* in_sizes, int n_in,
                              void* d_out, int out_size, void* d_ws,
                              size_t ws_size, hipStream_t stream) {
  const float* x = (const float*)d_in[0];
  const float* gamma = (const float*)d_in[1];
  const float* beta = (const float*)d_in[2];
  const float* centers = (const float*)d_in[3];
  const float* W = (const float*)d_in[4];
  const float* bias = (const float*)d_in[5];
  float* out = (float*)d_out;

  char* ws = (char*)d_ws;
  float* scale = (float*)(ws);
  float* shift = (float*)(ws + 4096);
  float* x_sq = (float*)(ws + 8192);
  float* c_sq = (float*)(ws + 24576);
  bf16* xn = (bf16*)(ws + 65536);
  bf16* ctT = (bf16*)(ws + 65536 + (size_t)8 * 1024 * 1024);
  bf16* Wb = (bf16*)(ws + 65536 + (size_t)12 * 1024 * 1024);
  bf16* phi = (bf16*)(ws + 65536 + (size_t)16 * 1024 * 1024);

  bn_stats_kernel<<<4, 256, 0, stream>>>(x, gamma, beta, scale, shift);
  csq_kernel<<<8, 256, 0, stream>>>(centers, c_sq);
  transpose_centers_kernel<<<dim3(CC / 32, II / 32), 256, 0, stream>>>(centers,
                                                                       ctT);
  castW_kernel<<<(OO * CC / 4) / 256, 256, 0, stream>>>(W, Wb);
  normalize_kernel<<<BB, 256, 0, stream>>>(x, scale, shift, xn, x_sq);
  gemm_phi_kernel<<<dim3(CC / 128, BB / 128), 256, 0, stream>>>(xn, ctT, x_sq,
                                                                c_sq, phi);
  gemm_out_kernel<<<dim3(OO / 128, BB / 128), 256, 0, stream>>>(phi, Wb, bias,
                                                                out);
}
